// DirectionalPropagation_18150531792934
// MI455X (gfx1250) — matrix operand order repaired and checked by reading, not compiled, not run on hardware
//
#include <hip/hip_runtime.h>
#include <hip/hip_bf16.h>
#include <math.h>

typedef __attribute__((ext_vector_type(16))) _Float16 v16h;
typedef __attribute__((ext_vector_type(8)))  float    v8f;

#define HID   256
#define TRANS 64
#define KDIM  512               // 2*HID after Wa/Wb fusion
#define N_DOM_C   20000
#define N_PROJ_C  100000
#define E_DSP_C   160000
#define E_DDOM_C  40000
#define E_PSP_C   1600000
#define E_PDOM_C  200000

// ---------------------------------------------------------------------------
// Kernel 0: x (f32) -> xh (f16), so staging gathers are pure 16B async copies.
// ---------------------------------------------------------------------------
__global__ void dp_x_f16(const float* __restrict__ x, _Float16* __restrict__ xh, int n) {
    int i = blockIdx.x * blockDim.x + threadIdx.x;
    if (i < n) xh[i] = (_Float16)x[i];
}

// ---------------------------------------------------------------------------
// Kernel 1: fuse W1 -> Wab = [W1a+W1c ; W1b-W1c] (512x256) as f16, pre-swizzled
// into WMMA B-fragment layout, and W2 (256x64) likewise.
// B fragment layout (16-bit B, 32x16): lane l holds column n = l%16;
// halves j=0..15 hold K = kstep*32 + (l<16 ? 0 : 8) + (j<8 ? j : 16 + j - 8).
// Fragment f stored as 512 contiguous halves: [lane*16 + j].
// Wab fragments: f = ntile*16 + kstep  (ntile<16, kstep<16)
// W2  fragments: f = ntile*8  + kstep  (ntile<4,  kstep<8)
// ---------------------------------------------------------------------------
__global__ void dp_prep_weights(const float* __restrict__ W1,
                                const float* __restrict__ W2,
                                _Float16* __restrict__ Wab,
                                _Float16* __restrict__ W2s) {
    int t = blockIdx.x * blockDim.x + threadIdx.x;
    if (t < 16 * 16 * 512) {           // Wab: 131072 halves
        int frag = t >> 9, within = t & 511;
        int lane = within >> 4, j = within & 15;
        int ntile = frag >> 4, kstep = frag & 15;
        int n = ntile * 16 + (lane & 15);
        int k = kstep * 32 + ((lane < 16) ? 0 : 8) + ((j < 8) ? j : 16 + (j - 8));
        float v;
        if (k < HID) v = W1[k * HID + n] + W1[(k + 2 * HID) * HID + n];     // Wa = W1a + W1c
        else         v = W1[k * HID + n] - W1[(k + HID) * HID + n];         // Wb = W1b - W1c
        Wab[t] = (_Float16)v;
    }
    if (t < 4 * 8 * 512) {             // W2s: 16384 halves
        int frag = t >> 9, within = t & 511;
        int lane = within >> 4, j = within & 15;
        int ntile = frag >> 3, kstep = frag & 7;
        int n = ntile * 16 + (lane & 15);
        int k = kstep * 32 + ((lane < 16) ? 0 : 8) + ((j < 8) ? j : 16 + (j - 8));
        W2s[t] = (_Float16)W2[k * TRANS + n];
    }
}

// ---------------------------------------------------------------------------
// Kernel 2: per-edge MLP via WMMA.
// Block = 256 thr (8 waves) handles 32 edges (two M-tiles).
//   A[32][512] f16 in LDS = [xh[src] | xh[dst]] rows, staged with
//   global_load_async_to_lds_b128 (ASYNCcnt path).
//   GEMM1: each wave -> 2 N-tiles of H = relu(A@Wab + b1); each B fragment is
//          reused across both M-tiles (2 WMMAs per B load).
//   GEMM2: all 8 waves (2 Mtiles x 4 Ntiles): trans = sigmoid(H@W2 + b2).
// ---------------------------------------------------------------------------
__global__ __launch_bounds__(256)
void dp_edge_trans(const _Float16* __restrict__ xh,
                   const int* __restrict__ src, const int* __restrict__ dst, int E,
                   const _Float16* __restrict__ Bsw, const float* __restrict__ b1,
                   const _Float16* __restrict__ B2sw, const float* __restrict__ b2,
                   float* __restrict__ trans) {
    __shared__ _Float16 A[32 * KDIM];    // 32 KB
    __shared__ _Float16 H[32 * HID];     // 16 KB
    const int tid = threadIdx.x;
    const int e0  = blockIdx.x * 32;

    // ---- stage A rows with async global->LDS 16B copies
    // 32 rows x 1024B = 2048 chunks of 16B; 8 chunks per thread.
    {
        unsigned ldsA = (unsigned)(uintptr_t)(&A[0]);
        #pragma unroll
        for (int it = 0; it < 8; ++it) {
            int c = tid + it * 256;          // chunk id 0..2047
            int m = c >> 6;                  // row (64 x 16B per row)
            int k = c & 63;                  // 16B chunk within row
            int e = e0 + m; if (e >= E) e = E - 1;
            int node = (k < 32) ? src[e] : dst[e];
            unsigned long long ga =
                (unsigned long long)(uintptr_t)xh + (unsigned long long)node * 512ull
                + (unsigned long long)((k & 31) * 16);
            unsigned la = ldsA + (unsigned)(m * 1024 + k * 16);
            asm volatile("global_load_async_to_lds_b128 %0, %1, off"
                         :: "v"(la), "v"(ga) : "memory");
        }
        asm volatile("s_wait_asynccnt 0x0" ::: "memory");
    }
    __syncthreads();

    const int w    = tid >> 5, lane = tid & 31;
    const int Mrow = lane & 15;
    const int kb   = (lane < 16) ? 0 : 8;      // A-fragment K base per ISA layout
    const int l16  = lane & 15;
    const int roff = (lane < 16) ? 0 : 8;      // C/D row offset per ISA layout

    // ---- GEMM1: H = relu(A @ Wab + b1); B reused across both M-tiles
    #pragma unroll
    for (int t2 = 0; t2 < 2; ++t2) {
        int nt = w * 2 + t2;
        v8f c0 = {}, c1 = {};
        #pragma unroll
        for (int kstep = 0; kstep < 16; ++kstep) {
            v16h b = *(const v16h*)(Bsw + ((size_t)(nt * 16 + kstep)) * 512 + lane * 16);
            const _Float16* ap0 = &A[Mrow * KDIM + kstep * 32 + kb];
            const _Float16* ap1 = &A[(Mrow + 16) * KDIM + kstep * 32 + kb];
            v16h a0, a1;
            #pragma unroll
            for (int i = 0; i < 8; ++i) {
                a0[i] = ap0[i]; a0[i + 8] = ap0[i + 16];
                a1[i] = ap1[i]; a1[i + 8] = ap1[i + 16];
            }
            c0 = __builtin_amdgcn_wmma_f32_16x16x32_f16(false, a0, false, b,
                                                        (short)0, c0, false, false);
            c1 = __builtin_amdgcn_wmma_f32_16x16x32_f16(false, a1, false, b,
                                                        (short)0, c1, false, false);
        }
        int n = nt * 16 + l16;
        float bias = b1[n];
        #pragma unroll
        for (int i = 0; i < 8; ++i) {
            float v0 = c0[i] + bias; v0 = v0 > 0.f ? v0 : 0.f;
            float v1 = c1[i] + bias; v1 = v1 > 0.f ? v1 : 0.f;
            H[(i + roff) * HID + n]      = (_Float16)v0;
            H[(i + roff + 16) * HID + n] = (_Float16)v1;
        }
    }
    __syncthreads();

    // ---- GEMM2: trans = sigmoid(H @ W2 + b2); all 8 waves (2 Mt x 4 Nt)
    {
        int mt = w >> 2;       // 0..1
        int nt = w & 3;        // 0..3
        v8f c = {};
        #pragma unroll
        for (int kstep = 0; kstep < 8; ++kstep) {
            const _Float16* ap = &H[(mt * 16 + Mrow) * HID + kstep * 32 + kb];
            v16h a;
            #pragma unroll
            for (int i = 0; i < 8; ++i) { a[i] = ap[i]; a[i + 8] = ap[i + 16]; }
            v16h b = *(const v16h*)(B2sw + ((size_t)(nt * 8 + kstep)) * 512 + lane * 16);
            c = __builtin_amdgcn_wmma_f32_16x16x32_f16(false, a, false, b,
                                                       (short)0, c, false, false);
        }
        int n = nt * 16 + l16;
        float bias = b2[n];
        #pragma unroll
        for (int i = 0; i < 8; ++i) {
            int e = e0 + mt * 16 + i + roff;
            if (e < E) {
                float v = c[i] + bias;
                trans[(size_t)e * TRANS + n] = 1.f / (1.f + __expf(-v));
            }
        }
    }
}

// ---------------------------------------------------------------------------
// Kernel 3: ew[e] = sigmoid(attr[e]·w[0:32] + trans[br[e]]·w[32:96] + b)
// ---------------------------------------------------------------------------
__global__ void dp_edge_weight(const float* __restrict__ attr,
                               const int* __restrict__ br,
                               const float* __restrict__ trans,
                               const float* __restrict__ wv,
                               const float* __restrict__ bp,
                               float* __restrict__ ew, int E) {
    int e = blockIdx.x * blockDim.x + threadIdx.x;
    if (e >= E) return;
    float acc = bp[0];
    const float4* a4 = (const float4*)(attr + (size_t)e * 32);
    #pragma unroll
    for (int i = 0; i < 8; ++i) {
        float4 v = a4[i];
        acc += v.x * wv[i*4+0] + v.y * wv[i*4+1] + v.z * wv[i*4+2] + v.w * wv[i*4+3];
    }
    const float4* t4 = (const float4*)(trans + (size_t)br[e] * TRANS);
    #pragma unroll
    for (int i = 0; i < 16; ++i) {
        float4 v = t4[i];
        acc += v.x * wv[32+i*4+0] + v.y * wv[32+i*4+1] + v.z * wv[32+i*4+2] + v.w * wv[32+i*4+3];
    }
    ew[e] = 1.f / (1.f + __expf(-acc));
}

// ---------------------------------------------------------------------------
// APPNP helpers
// ---------------------------------------------------------------------------
__global__ void dp_fill(float* __restrict__ p, float v, int N) {
    int i = blockIdx.x * blockDim.x + threadIdx.x;
    if (i < N) p[i] = v;
}
__global__ void dp_relu(const float* __restrict__ a, float* __restrict__ o, int N) {
    int i = blockIdx.x * blockDim.x + threadIdx.x;
    if (i < N) o[i] = fmaxf(a[i], 0.f);
}
__global__ void dp_deg_add(float* __restrict__ deg, const int* __restrict__ col,
                           const float* __restrict__ ew, int E) {
    int e = blockIdx.x * blockDim.x + threadIdx.x;
    if (e < E) atomicAdd(&deg[col[e]], ew[e]);
}
__global__ void dp_rsqrt_inplace(float* __restrict__ p, int N) {
    int i = blockIdx.x * blockDim.x + threadIdx.x;
    if (i < N) { float d = p[i]; p[i] = d > 0.f ? rsqrtf(d) : 0.f; }
}
// h_next = alpha*h0 + (1-alpha) * dinv^2 * h_cur   (self-loop, w=1)
__global__ void dp_appnp_node(float* __restrict__ hn, const float* __restrict__ h0,
                              const float* __restrict__ hc, const float* __restrict__ dinv,
                              const float* __restrict__ alphap, int N) {
    int i = blockIdx.x * blockDim.x + threadIdx.x;
    if (i >= N) return;
    float al = alphap[0], di = dinv[i];
    hn[i] = al * h0[i] + (1.f - al) * di * di * hc[i];
}
// h_next[col] += (1-alpha) * dinv[row]*ew*dinv[col] * h_cur[row]
__global__ void dp_appnp_edge(float* __restrict__ hn, const float* __restrict__ hc,
                              const float* __restrict__ dinv, const float* __restrict__ ew,
                              const int* __restrict__ row, const int* __restrict__ col,
                              const float* __restrict__ alphap, int E) {
    int e = blockIdx.x * blockDim.x + threadIdx.x;
    if (e >= E) return;
    float al = alphap[0];
    int r = row[e], c = col[e];
    atomicAdd(&hn[c], (1.f - al) * dinv[r] * ew[e] * dinv[c] * hc[r]);
}
// out = max(mask, max(tanh(ph - sp(pb)), tanh(dh - sp(db))))
__global__ void dp_final(const float* __restrict__ mask,
                         const float* __restrict__ ph, const float* __restrict__ pbias,
                         const float* __restrict__ dh, const float* __restrict__ dbias,
                         float* __restrict__ out, int N) {
    int i = blockIdx.x * blockDim.x + threadIdx.x;
    if (i >= N) return;
    float spp = logf(1.f + __expf(pbias[0]));
    float spd = logf(1.f + __expf(dbias[0]));
    float pv = tanhf(ph[i] - spp);
    float dv = tanhf(dh[i] - spd);
    out[i] = fmaxf(mask[i], fmaxf(pv, dv));
}

// ---------------------------------------------------------------------------
extern "C" void kernel_launch(void* const* d_in, const int* in_sizes, int n_in,
                              void* d_out, int out_size, void* d_ws, size_t ws_size,
                              hipStream_t stream) {
    const float* x        = (const float*)d_in[0];
    const float* mask     = (const float*)d_in[1];
    const int*   dsp_ei   = (const int*)d_in[2];   // [2, E_DSP]
    const int*   ddom_ei  = (const int*)d_in[3];   // [2, E_DDOM]
    const int*   psp_ei   = (const int*)d_in[4];   // [2, E_PSP]
    const int*   psp_br   = (const int*)d_in[5];
    const float* psp_attr = (const float*)d_in[6];
    const int*   pdom_ei  = (const int*)d_in[7];   // [2, E_PDOM]
    const int*   pdom_br  = (const int*)d_in[8];
    const float* pdom_attr= (const float*)d_in[9];
    const float* W1 = (const float*)d_in[10];
    const float* b1 = (const float*)d_in[11];
    const float* W2 = (const float*)d_in[12];
    const float* b2 = (const float*)d_in[13];
    const float* pos_w = (const float*)d_in[14];
    const float* pos_b = (const float*)d_in[15];
    const float* pos_alpha = (const float*)d_in[16];
    const float* pos_bias  = (const float*)d_in[17];
    const float* dom_w = (const float*)d_in[18];
    const float* dom_b = (const float*)d_in[19];
    const float* dom_alpha = (const float*)d_in[20];
    const float* dom_bias  = (const float*)d_in[21];
    float* out = (float*)d_out;

    // workspace carve-up (256B aligned)
    size_t off = 0;
    char* base = (char*)d_ws;
    auto carve = [&](size_t bytes) -> char* {
        char* p = base + off;
        off += (bytes + 255) & ~(size_t)255;
        return p;
    };
    _Float16* xh  = (_Float16*)carve((size_t)N_DOM_C * HID * sizeof(_Float16));
    _Float16* Wab = (_Float16*)carve(16 * 16 * 512 * sizeof(_Float16));
    _Float16* W2s = (_Float16*)carve(4 * 8 * 512 * sizeof(_Float16));
    float* trans_sp  = (float*)carve((size_t)E_DSP_C  * TRANS * 4);
    float* trans_dom = (float*)carve((size_t)E_DDOM_C * TRANS * 4);
    float* ew_sp  = (float*)carve((size_t)E_PSP_C  * 4);
    float* ew_dom = (float*)carve((size_t)E_PDOM_C * 4);
    float* dinv = (float*)carve((size_t)N_PROJ_C * 4);   // deg -> dinv in place
    float* h0   = (float*)carve((size_t)N_PROJ_C * 4);
    float* pA   = (float*)carve((size_t)N_PROJ_C * 4);
    float* pB   = (float*)carve((size_t)N_PROJ_C * 4);
    float* dA   = (float*)carve((size_t)N_PROJ_C * 4);
    float* dB   = (float*)carve((size_t)N_PROJ_C * 4);

    const int T = 256;
    auto gs = [&](long long n) { return (int)((n + T - 1) / T); };

    // 0) x -> f16
    dp_x_f16<<<gs((long long)N_DOM_C * HID), T, 0, stream>>>(x, xh, N_DOM_C * HID);

    // 1) fused + swizzled weights
    dp_prep_weights<<<gs(16 * 16 * 512), T, 0, stream>>>(W1, W2, Wab, W2s);

    // 2) edge-MLP via WMMA (spatial + dom edge sets), 32 edges / block
    dp_edge_trans<<<E_DSP_C / 32, T, 0, stream>>>(
        xh, dsp_ei, dsp_ei + E_DSP_C, E_DSP_C, Wab, b1, W2s, b2, trans_sp);
    dp_edge_trans<<<E_DDOM_C / 32, T, 0, stream>>>(
        xh, ddom_ei, ddom_ei + E_DDOM_C, E_DDOM_C, Wab, b1, W2s, b2, trans_dom);

    // 3) projected edge weights
    dp_edge_weight<<<gs(E_PSP_C), T, 0, stream>>>(
        psp_attr, psp_br, trans_sp, pos_w, pos_b, ew_sp, E_PSP_C);
    dp_edge_weight<<<gs(E_PDOM_C), T, 0, stream>>>(
        pdom_attr, pdom_br, trans_dom, dom_w, dom_b, ew_dom, E_PDOM_C);

    // 4) shared restart vector h0 = relu(mask)
    dp_relu<<<gs(N_PROJ_C), T, 0, stream>>>(mask, h0, N_PROJ_C);

    // 5) APPNP pipeline, run twice (pos, dom)
    auto appnp = [&](const int* row, const int* col, const float* ew, int E,
                     const float* alphap, float* bufA, float* bufB) {
        dp_fill<<<gs(N_PROJ_C), T, 0, stream>>>(dinv, 1.0f, N_PROJ_C);      // self-loop deg
        dp_deg_add<<<gs(E), T, 0, stream>>>(dinv, col, ew, E);
        dp_rsqrt_inplace<<<gs(N_PROJ_C), T, 0, stream>>>(dinv, N_PROJ_C);
        const float* cur = h0;
        float* nxt = bufA;
        for (int k = 0; k < 5; ++k) {
            dp_appnp_node<<<gs(N_PROJ_C), T, 0, stream>>>(nxt, h0, cur, dinv, alphap, N_PROJ_C);
            dp_appnp_edge<<<gs(E), T, 0, stream>>>(nxt, cur, dinv, ew, row, col, alphap, E);
            cur = nxt;
            nxt = (cur == bufA) ? bufB : bufA;
        }
        return (float*)cur;   // 5 steps -> bufA
    };
    float* pos_h = appnp(psp_ei, psp_ei + E_PSP_C, ew_sp, E_PSP_C, pos_alpha, pA, pB);
    float* dom_h = appnp(pdom_ei, pdom_ei + E_PDOM_C, ew_dom, E_PDOM_C, dom_alpha, dA, dB);

    // 6) out = max(mask, max(tanh(pos - sp), tanh(dom - sp)))
    dp_final<<<gs(N_PROJ_C), T, 0, stream>>>(mask, pos_h, pos_bias, dom_h, dom_bias,
                                             out, N_PROJ_C);
    (void)in_sizes; (void)n_in; (void)out_size; (void)ws_size;
}
